// SelfAttention_45329084842444
// MI455X (gfx1250) — matrix operand order repaired and checked by reading, not compiled, not run on hardware
//
#include <hip/hip_runtime.h>
#include <stdint.h>

#define B_ 4
#define C_ 256
#define N_ 4096
#define D_ 32

typedef __bf16 bf16;
typedef __attribute__((ext_vector_type(8)))  bf16  v8bf;
typedef __attribute__((ext_vector_type(16))) bf16  v16bf;
typedef __attribute__((ext_vector_type(8)))  float v8f;

// Concatenate two contiguous 8x bf16 chunks into a WMMA 16-bf16 operand.
__device__ __forceinline__ v16bf load2x8(const bf16* p0, const bf16* p1) {
  v8bf lo = *(const v8bf*)p0;
  v8bf hi = *(const v8bf*)p1;
  v16bf r;
#pragma unroll
  for (int i = 0; i < 8; ++i) { r[i] = lo[i]; r[i + 8] = hi[i]; }
  return r;
}

__device__ __forceinline__ void store_bf8(bf16* p, const float* f) {
  v8bf t;
#pragma unroll
  for (int i = 0; i < 8; ++i) t[i] = (bf16)f[i];
  *(v8bf*)p = t;
}

// ---------------------------------------------------------------------------
// K1: q/k projections. qT/kT stored [b][n][32] bf16 (d contiguous) so WMMA
// operand loads are contiguous 16B per lane.
// ---------------------------------------------------------------------------
__global__ __launch_bounds__(256) void proj_qk_kernel(
    const float* __restrict__ x,
    const float* __restrict__ Wq, const float* __restrict__ bq,
    const float* __restrict__ Wk, const float* __restrict__ bk,
    bf16* __restrict__ qT, bf16* __restrict__ kT)
{
  int blk = blockIdx.x;                      // B * 16 blocks
  int b = blk >> 4;
  int n = ((blk & 15) << 8) + threadIdx.x;
  const float* xb = x + (size_t)b * C_ * N_ + n;

  float aq[D_], ak[D_];
#pragma unroll
  for (int d = 0; d < D_; ++d) { aq[d] = bq[d]; ak[d] = bk[d]; }

  for (int c = 0; c < C_; ++c) {
    float xv = xb[(size_t)c * N_];
#pragma unroll
    for (int d = 0; d < D_; ++d) {
      aq[d] = fmaf(Wq[d * C_ + c], xv, aq[d]);   // W broadcast -> SGPR
      ak[d] = fmaf(Wk[d * C_ + c], xv, ak[d]);
    }
  }
  size_t base = ((size_t)b * N_ + n) * D_;
#pragma unroll
  for (int i = 0; i < 4; ++i) {
    store_bf8(qT + base + i * 8, aq + i * 8);
    store_bf8(kT + base + i * 8, ak + i * 8);
  }
}

// ---------------------------------------------------------------------------
// K2: v projection, 32 output channels per thread (8x less x re-read).
// v stored [b][c][n] bf16 (n contiguous) = WMMA A layout for the out-GEMM.
// ---------------------------------------------------------------------------
__global__ __launch_bounds__(256) void proj_v_kernel(
    const float* __restrict__ x, const float* __restrict__ Wv,
    const float* __restrict__ bv, bf16* __restrict__ vB)
{
  int blk = blockIdx.x;                      // B * 16 * 8 = 512 blocks
  int b = blk >> 7;
  int rem = blk & 127;
  int c0 = (rem >> 4) << 5;                  // 8 groups of 32 channels
  int n = ((rem & 15) << 8) + threadIdx.x;

  const float* xb = x + (size_t)b * C_ * N_ + n;
  float acc[32];
#pragma unroll
  for (int i = 0; i < 32; ++i) acc[i] = bv[c0 + i];

  for (int cc = 0; cc < C_; ++cc) {
    float xv = xb[(size_t)cc * N_];
#pragma unroll
    for (int i = 0; i < 32; ++i)
      acc[i] = fmaf(Wv[(c0 + i) * C_ + cc], xv, acc[i]);  // W -> SGPR
  }
  bf16* vp = vB + ((size_t)b * C_ + c0) * N_ + n;
#pragma unroll
  for (int i = 0; i < 32; ++i) vp[(size_t)i * N_] = (bf16)acc[i];
}

// ---------------------------------------------------------------------------
// K3: fused attention for one 16-row query strip.
//   Pass 1: energy tiles via WMMA (discarded), online per-row (max, sum-exp).
//   Pass 2: recompute energy tiles via WMMA, normalize in registers, write
//           final attention fp32 ONCE, stage bf16 att^T in LDS, and run the
//           out-GEMM out[c,i] = sum_j v[c,j]*att[i,j] with WMMA (K=j).
//   Epilogue: out = gamma*out + x.
// Attention tensor HBM traffic: single 268 MB write (no raw-score round trip).
// ---------------------------------------------------------------------------
__global__ __launch_bounds__(256) void attn_fused_kernel(
    const bf16* __restrict__ qT, const bf16* __restrict__ kT,
    const bf16* __restrict__ vB, const float* __restrict__ x,
    const float* __restrict__ gamma,
    float* __restrict__ att, float* __restrict__ out)
{
  __shared__ __align__(16) bf16 ldsB[8][16][32];   // att[i][j] staging, 8 KB
  __shared__ float pM[16][128], pS[16][128];       // online-softmax partials
  __shared__ float rowM[16], rowIS[16];

  int blk = blockIdx.x;                      // B * 256 blocks
  int b = blk >> 8;
  int i0 = (blk & 255) << 4;
  int tid = threadIdx.x;
  int wave = tid >> 5, lane = tid & 31, h = lane >> 4, lr = lane & 15;

  const bf16* kTb = kT + (size_t)b * N_ * D_;

  // A = q^T tile (16 rows i, K=32): held in registers for both passes.
  const bf16* qrow = qT + ((size_t)b * N_ + (i0 + lr)) * D_ + h * 8;
  v16bf A = load2x8(qrow, qrow + 16);

  // ---- Pass 1: online (max, sum-exp) over all N columns, energy by WMMA ----
  {
    float m[8], s[8];
#pragma unroll
    for (int r = 0; r < 8; ++r) { m[r] = -3.0e38f; s[r] = 0.f; }

    for (int jt = wave; jt < N_ / 16; jt += 8) {
      int j0 = jt << 4;
      const bf16* krow = kTb + (size_t)(j0 + lr) * D_ + h * 8;
      v16bf Bm = load2x8(krow, krow + 16);
      v8f e = {0.f, 0.f, 0.f, 0.f, 0.f, 0.f, 0.f, 0.f};
      e = __builtin_amdgcn_wmma_f32_16x16x32_bf16(
          false, A, false, Bm, (short)0, e, false, false);
#pragma unroll
      for (int r = 0; r < 8; ++r) {
        float nm = fmaxf(m[r], e[r]);
        s[r] = s[r] * __expf(m[r] - nm) + __expf(e[r] - nm);
        m[r] = nm;
      }
    }
#pragma unroll
    for (int r = 0; r < 8; ++r) {            // D row r lives at matrix row r+8h
      pM[r + 8 * h][wave * 16 + lr] = m[r];
      pS[r + 8 * h][wave * 16 + lr] = s[r];
    }
  }
  __syncthreads();
  if (tid < 16) {                            // combine 128 partials per row
    float m = -3.0e38f;
    for (int i = 0; i < 128; ++i) m = fmaxf(m, pM[tid][i]);
    float s = 0.f;
    for (int i = 0; i < 128; ++i) s += pS[tid][i] * __expf(pM[tid][i] - m);
    rowM[tid] = m;
    rowIS[tid] = 1.0f / s;
  }
  __syncthreads();

  // per-lane copies of the stats for the 8 rows this lane's D registers hold
  float rm[8], ris[8];
#pragma unroll
  for (int r = 0; r < 8; ++r) { rm[r] = rowM[r + 8 * h]; ris[r] = rowIS[r + 8 * h]; }

  // ---- Pass 2: recompute, normalize, emit attention + out-GEMM ----
  int c0 = wave << 5;                        // 32 output channels per wave
  v8f d0 = {0.f, 0.f, 0.f, 0.f, 0.f, 0.f, 0.f, 0.f};
  v8f d1 = d0;
  const bf16* vb = vB + (size_t)b * C_ * N_;
  float* attw = att + (size_t)b * N_ * N_ + (size_t)i0 * N_;

  for (int ot = 0; ot < 16; ++ot) {
    int jbase = ot << 8;                     // 256 columns per outer step
    int j0w = jbase + (wave << 5);           // this wave's 32 columns
#pragma unroll
    for (int t = 0; t < 2; ++t) {            // two 16x16 energy tiles
      int j0 = j0w + (t << 4);
      const bf16* krow = kTb + (size_t)(j0 + lr) * D_ + h * 8;
      v16bf Bm = load2x8(krow, krow + 16);
      v8f e = {0.f, 0.f, 0.f, 0.f, 0.f, 0.f, 0.f, 0.f};
      e = __builtin_amdgcn_wmma_f32_16x16x32_bf16(
          false, A, false, Bm, (short)0, e, false, false);
#pragma unroll
      for (int r = 0; r < 8; ++r) {
        float p = __expf(e[r] - rm[r]) * ris[r];
        attw[(size_t)(r + 8 * h) * N_ + (j0 + lr)] = p;      // final attention
        ldsB[wave][r + 8 * h][(t << 4) + lr] = (bf16)p;      // B-operand stage
      }
    }
    __syncthreads();

#pragma unroll
    for (int sub = 0; sub < 8; ++sub) {
      int j0 = jbase + (sub << 5);
      // B = att^T (K=32 rows j, 16 cols i): lane reads att[lr] chunks {8h, 8h+16}
      const bf16* lp = &ldsB[sub][lr][h << 3];
      v16bf Bm = load2x8(lp, lp + 16);
      // A = v tiles (16 c-rows, K=32 j): contiguous chunks {8h, 8h+16}
      const bf16* vr0 = vb + (size_t)(c0 + lr) * N_ + j0 + h * 8;
      v16bf A0 = load2x8(vr0, vr0 + 16);
      const bf16* vr1 = vb + (size_t)(c0 + 16 + lr) * N_ + j0 + h * 8;
      v16bf A1 = load2x8(vr1, vr1 + 16);
      d0 = __builtin_amdgcn_wmma_f32_16x16x32_bf16(
          false, A0, false, Bm, (short)0, d0, false, false);
      d1 = __builtin_amdgcn_wmma_f32_16x16x32_bf16(
          false, A1, false, Bm, (short)0, d1, false, false);
    }
    __syncthreads();
  }

  // ---- Epilogue: out = gamma*out + x ----
  float g = gamma[0];
  const float* xb = x + (size_t)b * C_ * N_;
  float* ob = out + (size_t)b * C_ * N_;
#pragma unroll
  for (int r = 0; r < 8; ++r) {
    int c = c0 + r + h * 8;
    size_t idx0 = (size_t)c * N_ + (i0 + lr);
    size_t idx1 = (size_t)(c + 16) * N_ + (i0 + lr);
    ob[idx0] = g * d0[r] + xb[idx0];
    ob[idx1] = g * d1[r] + xb[idx1];
  }
}

// ---------------------------------------------------------------------------
extern "C" void kernel_launch(void* const* d_in, const int* in_sizes, int n_in,
                              void* d_out, int out_size, void* d_ws, size_t ws_size,
                              hipStream_t stream) {
  const float* x     = (const float*)d_in[0];
  const float* Wq    = (const float*)d_in[1];
  const float* bq    = (const float*)d_in[2];
  const float* Wk    = (const float*)d_in[3];
  const float* bk    = (const float*)d_in[4];
  const float* Wv    = (const float*)d_in[5];
  const float* bv    = (const float*)d_in[6];
  const float* gamma = (const float*)d_in[7];

  float* out = (float*)d_out;                          // [B,C,N]
  float* att = out + (size_t)B_ * C_ * N_;             // [B,N,N]

  bf16* qT = (bf16*)d_ws;                              // [B,N,32]  1 MB
  bf16* kT = qT + (size_t)B_ * N_ * D_;                // [B,N,32]  1 MB
  bf16* vB = kT + (size_t)B_ * N_ * D_;                // [B,C,N]   8 MB

  proj_qk_kernel<<<B_ * (N_ / 256), 256, 0, stream>>>(x, Wq, bq, Wk, bk, qT, kT);
  proj_v_kernel<<<B_ * 128, 256, 0, stream>>>(x, Wv, bv, vB);
  attn_fused_kernel<<<B_ * (N_ / 16), 256, 0, stream>>>(qT, kT, vB, x, gamma, att, out);
}
